// InvariantPointAttention_30090540875810
// MI455X (gfx1250) — compile-verified, matrix operand order checked by reading, not run on hardware
//
#include <hip/hip_runtime.h>
#include <hip/hip_bf16.h>

typedef __attribute__((ext_vector_type(16))) _Float16 v16h;
typedef __attribute__((ext_vector_type(8)))  _Float16 v8h;
typedef __attribute__((ext_vector_type(8)))  float    v8f;
typedef __attribute__((ext_vector_type(4))) unsigned int v4u;
typedef __attribute__((ext_vector_type(4))) int v4i;
typedef __attribute__((ext_vector_type(8))) int v8i;

#if __has_builtin(__builtin_amdgcn_tensor_load_to_lds) && __has_builtin(__builtin_amdgcn_s_wait_tensorcnt)
#define HAVE_TDM 1
#else
#define HAVE_TDM 0
#endif

namespace {
constexpr int kB = 2, kN = 512, kDIM = 384, kH = 12;
constexpr int kPAIR = 128;
constexpr int kCOL = 2112;                               // H*(16 + 8*4 + 128)
constexpr float kScalarScale = 0.14433756729740643f;     // (3*16)^-0.5
constexpr float kPointScale  = 0.13608276348795434f;     // (3*4*4.5)^-0.5
constexpr float kPairScale   = 0.57735026918962576f;     // 3^-0.5
constexpr float kEps = 1e-8f;
}

// ---------------- WMMA + fragment helpers (ISA 7.12.2 layouts) --------------

__device__ __forceinline__ v8f wmma32(v16h a, v16h b, v8f c) {
  return __builtin_amdgcn_wmma_f32_16x16x32_f16(false, a, false, b, (short)0, c,
                                                false, false);
}

// Fragment from f16 row: halves [k+kb .. k+kb+7] and [k+kb+16 .. k+kb+23].
__device__ __forceinline__ v16h fragH(const _Float16* p) {
  v16h f;
#pragma unroll
  for (int e = 0; e < 8; ++e) { f[e] = p[e]; f[e + 8] = p[e + 16]; }
  return f;
}

// Fragment from f32 row (converted): p = rowBase + k + kb.
__device__ __forceinline__ v16h fragF32row(const float* p) {
  v16h f;
#pragma unroll
  for (int e = 0; e < 8; ++e) {
    f[e]     = (_Float16)p[e];
    f[e + 8] = (_Float16)p[e + 16];
  }
  return f;
}

#if HAVE_TDM
// Issue a TDM load of one contiguous 32x128 f32 chunk (16 KB) into LDS.
__device__ __forceinline__ void tdm_load_chunk(const float* src, unsigned ldsOff) {
  unsigned long long ga = (unsigned long long)(const void*)src;
  v4u g0;
  g0[0] = 1u;                                   // count=1 (valid user D#)
  g0[1] = ldsOff;                               // lds_addr (bytes)
  g0[2] = (unsigned)(ga & 0xffffffffu);         // global_addr[31:0]
  g0[3] = (unsigned)(ga >> 32) | (2u << 30);    // global_addr[56:32] | type=2
  v8i g1;
  g1[0] = 0x20000;                              // data_size=4B; no mask/pad/iter
  g1[1] = (int)(4096u << 16);                   // tensor_dim0 = 4096 elements
  g1[2] = (int)(1u << 16);                      // tensor_dim1 = 1
  g1[3] = (int)(4096u << 16);                   // tile_dim0 = 4096 (1-D tile)
  g1[4] = 0;                                    // tile_dim1/2 unused
  g1[5] = 4096;                                 // tensor_dim0_stride
  g1[6] = 0; g1[7] = 0;
  v4i gz = {0, 0, 0, 0};
#if defined(__clang_major__) && __clang_major__ >= 23
  v8i gz8 = {0, 0, 0, 0, 0, 0, 0, 0};
  __builtin_amdgcn_tensor_load_to_lds(g0, g1, gz, gz, gz8, 0);
#else
  __builtin_amdgcn_tensor_load_to_lds(g0, g1, gz, gz, 0);
#endif
}
#endif

// ---------------- Conversion kernels ----------------------------------------

__global__ void k_cvt_h16(const float* __restrict__ src, _Float16* __restrict__ dst, int n) {
  int i = blockIdx.x * blockDim.x + threadIdx.x;
  if (i < n) dst[i] = (_Float16)src[i];
}

// W (R x C, f32, row-major) -> Wt (Cpad x R, f16, row-major); rows >= C zeroed.
__global__ void k_cvt_wt(const float* __restrict__ W, int R, int C,
                         _Float16* __restrict__ Wt, int Cpad) {
  int i = blockIdx.x * blockDim.x + threadIdx.x;
  if (i >= Cpad * R) return;
  int c = i / R, r = i % R;
  Wt[(long)c * R + r] = (c < C) ? (_Float16)W[(long)r * C + c] : (_Float16)0.f;
}

// ---------------- GEMM: C(f32) = A(f16,MxK) * Bt(f16,NxK)^T (+bias) ---------
__global__ void k_gemm_n(const _Float16* __restrict__ A, int lda,
                         const _Float16* __restrict__ Bt, int ldbt,
                         float* __restrict__ C, int ldc,
                         int K, int Nn, const float* __restrict__ bias) {
  int lane = threadIdx.x;
  int m0 = blockIdx.x * 16;
  int n0 = (blockIdx.y * 4 + threadIdx.y) * 16;
  if (n0 >= Nn) return;
  int r = lane & 15, kb = (lane < 16) ? 0 : 8;
  const _Float16* Ar = A + (long)(m0 + r) * lda + kb;
  const _Float16* Br = Bt + (long)(n0 + r) * ldbt + kb;
  v8f acc = {};
  for (int k = 0; k < K; k += 32)
    acc = wmma32(fragH(Ar + k), fragH(Br + k), acc);
  int col = n0 + r, mb = (lane < 16) ? 0 : 8;
  float bv = bias ? bias[col] : 0.f;
#pragma unroll
  for (int q = 0; q < 8; ++q)
    C[(long)(m0 + mb + q) * ldc + col] = acc[q] + bv;
}

// Same GEMM but stores C transposed as f16: Ct[col][m] (contiguous 16B store).
__global__ void k_gemm_t(const _Float16* __restrict__ A, int lda,
                         const _Float16* __restrict__ Bt, int ldbt,
                         _Float16* __restrict__ Ct, int ldct, int K, int Nn) {
  int lane = threadIdx.x;
  int m0 = blockIdx.x * 16;
  int n0 = (blockIdx.y * 4 + threadIdx.y) * 16;
  if (n0 >= Nn) return;
  int r = lane & 15, kb = (lane < 16) ? 0 : 8;
  const _Float16* Ar = A + (long)(m0 + r) * lda + kb;
  const _Float16* Br = Bt + (long)(n0 + r) * ldbt + kb;
  v8f acc = {};
  for (int k = 0; k < K; k += 32)
    acc = wmma32(fragH(Ar + k), fragH(Br + k), acc);
  int col = n0 + r, mb = (lane < 16) ? 0 : 8;
  v8h pk;
#pragma unroll
  for (int q = 0; q < 8; ++q) pk[q] = (_Float16)acc[q];
  *(v8h*)(Ct + (long)col * ldct + m0 + mb) = pk;
}

// ---------------- Rotate points, pack QK, norm adds, v_p (transposed f16) ---
__global__ void k_rotate_pack(const float* __restrict__ sq, const float* __restrict__ sk,
                              const float* __restrict__ pq, const float* __restrict__ pk,
                              const float* __restrict__ pv,
                              const float* __restrict__ rot, const float* __restrict__ trans,
                              const float* __restrict__ pw_raw,
                              _Float16* __restrict__ Ap, _Float16* __restrict__ Bp,
                              float* __restrict__ rowAdd, float* __restrict__ colAdd,
                              _Float16* __restrict__ vpT) {
  int idx = blockIdx.x * blockDim.x + threadIdx.x;
  if (idx >= kB * kN * kH) return;
  int h = idx % kH, bn = idx / kH;
  int b = bn / kN, i = bn % kN;
  const float* R = rot + (long)bn * 9;
  const float* t = trans + (long)bn * 3;
  float pw = log1pf(expf(pw_raw[h]));
  float pscale = pw * kPointScale;

  _Float16* ap = Ap + (((long)b * kH + h) * kN + i) * 32;
  _Float16* bp = Bp + (((long)b * kH + h) * kN + i) * 32;
  const float* qs = sq + (long)bn * 192 + h * 16;
  const float* ks = sk + (long)bn * 192 + h * 16;
#pragma unroll
  for (int d = 0; d < 16; ++d) {
    ap[d] = (_Float16)(qs[d] * kScalarScale);
    bp[d] = (_Float16)ks[d];
  }
  float qn2 = 0.f, kn2 = 0.f;
  const float* qp = pq + (long)bn * 144 + h * 12;
  const float* kp = pk + (long)bn * 144 + h * 12;
#pragma unroll
  for (int d = 0; d < 4; ++d) {
    float x = qp[d*3], y = qp[d*3+1], z = qp[d*3+2];
    float r0 = x*R[0] + y*R[3] + z*R[6] + t[0];   // out[r] = sum_c p[c]*R[c,r]
    float r1 = x*R[1] + y*R[4] + z*R[7] + t[1];
    float r2 = x*R[2] + y*R[5] + z*R[8] + t[2];
    qn2 += r0*r0 + r1*r1 + r2*r2;
    ap[16+d*3+0] = (_Float16)(r0*pscale);
    ap[16+d*3+1] = (_Float16)(r1*pscale);
    ap[16+d*3+2] = (_Float16)(r2*pscale);
    x = kp[d*3]; y = kp[d*3+1]; z = kp[d*3+2];
    r0 = x*R[0] + y*R[3] + z*R[6] + t[0];
    r1 = x*R[1] + y*R[4] + z*R[7] + t[1];
    r2 = x*R[2] + y*R[5] + z*R[8] + t[2];
    kn2 += r0*r0 + r1*r1 + r2*r2;
    bp[16+d*3+0] = (_Float16)r0;
    bp[16+d*3+1] = (_Float16)r1;
    bp[16+d*3+2] = (_Float16)r2;
  }
#pragma unroll
  for (int d = 28; d < 32; ++d) { ap[d] = (_Float16)0.f; bp[d] = (_Float16)0.f; }
  rowAdd[((long)b * kH + h) * kN + i] = -0.5f * pw * kPointScale * qn2;
  colAdd[((long)b * kH + h) * kN + i] = -0.5f * pw * kPointScale * kn2;

  const float* vp = pv + (long)bn * 288 + h * 24;
#pragma unroll
  for (int d = 0; d < 8; ++d) {
    float x = vp[d*3], y = vp[d*3+1], z = vp[d*3+2];
    float o0 = x*R[0] + y*R[3] + z*R[6] + t[0];
    float o1 = x*R[1] + y*R[4] + z*R[7] + t[1];
    float o2 = x*R[2] + y*R[5] + z*R[8] + t[2];
    long base = ((long)h * 24 + d * 3) * (kB * kN) + (long)b * kN + i;
    vpT[base]                 = (_Float16)o0;
    vpT[base + (kB * kN)]     = (_Float16)o1;
    vpT[base + 2 * (kB * kN)] = (_Float16)o2;
  }
}

// ---------------- Pair bias: logits[b,h,i,j] = (pair@W_pb + b)*PAIR_SCALE ---
__global__ void k_pair_bias(const float* __restrict__ pair,
                            const _Float16* __restrict__ WpbT,   // 16 x 128
                            const float* __restrict__ bpb,
                            float* __restrict__ logits) {
  int lane = threadIdx.x;
  int bi = blockIdx.x, b = bi / kN, i = bi % kN;
  int j0 = (blockIdx.y * 4 + threadIdx.y) * 16;
  int r = lane & 15, kb = (lane < 16) ? 0 : 8;
  const float* Ar = pair + ((long)bi * kN + j0 + r) * kPAIR + kb;
  const _Float16* Br = WpbT + (long)r * kPAIR + kb;
  v8f acc = {};
#pragma unroll
  for (int k = 0; k < kPAIR; k += 32)
    acc = wmma32(fragF32row(Ar + k), fragH(Br + k), acc);
  int c = r, mb = (lane < 16) ? 0 : 8;
  if (c < kH) {
    float bb = bpb[c];
    float* L = logits + (((long)b * kH + c) * kN + i) * kN;
#pragma unroll
    for (int q = 0; q < 8; ++q)
      L[j0 + mb + q] = (acc[q] + bb) * kPairScale;
  }
}

// ---------------- QK + point logits: C operand = pair bias ------------------
__global__ void k_qk_logits(const _Float16* __restrict__ Ap,
                            const _Float16* __restrict__ Bp,
                            const float* __restrict__ rowAdd,
                            const float* __restrict__ colAdd,
                            float* __restrict__ logits) {
  int lane = threadIdx.x;
  int bh = blockIdx.x;
  int i0 = blockIdx.y * 16;
  int j0 = (blockIdx.z * 4 + threadIdx.y) * 16;
  int r = lane & 15, kb = (lane < 16) ? 0 : 8;
  v16h a = fragH(Ap + ((long)bh * kN + i0 + r) * 32 + kb);
  v16h b = fragH(Bp + ((long)bh * kN + j0 + r) * 32 + kb);
  int c = r, mb = (lane < 16) ? 0 : 8;
  float* L = logits + ((long)bh * kN + i0) * kN + j0;
  v8f acc;
#pragma unroll
  for (int q = 0; q < 8; ++q) acc[q] = L[(long)(mb + q) * kN + c];
  acc = wmma32(a, b, acc);
  float ca = colAdd[(long)bh * kN + j0 + c];
#pragma unroll
  for (int q = 0; q < 8; ++q)
    L[(long)(mb + q) * kN + c] = acc[q] + rowAdd[(long)bh * kN + i0 + mb + q] + ca;
}

// ---------------- Softmax: one wave per (b,h,i) row, f16 probs out ----------
__global__ void k_softmax(const float* __restrict__ logits, _Float16* __restrict__ attn) {
  long row = (long)blockIdx.x * 4 + threadIdx.y;
  const float* L = logits + row * kN;
  _Float16* O = attn + row * kN;
  int lane = threadIdx.x;
  float v[16], mx = -3.4e38f;
#pragma unroll
  for (int e = 0; e < 16; ++e) { v[e] = L[lane + e * 32]; mx = fmaxf(mx, v[e]); }
#pragma unroll
  for (int off = 16; off >= 1; off >>= 1) mx = fmaxf(mx, __shfl_xor(mx, off));
  float s = 0.f;
#pragma unroll
  for (int e = 0; e < 16; ++e) { v[e] = expf(v[e] - mx); s += v[e]; }
#pragma unroll
  for (int off = 16; off >= 1; off >>= 1) s += __shfl_xor(s, off);
  float inv = 1.f / s;
#pragma unroll
  for (int e = 0; e < 16; ++e) O[lane + e * 32] = (_Float16)(v[e] * inv);
}

// ---------------- attn(f16) @ Vt(f16): per (b,h), M=512, K=512 --------------
__global__ void k_attn_v(const _Float16* __restrict__ attn,
                         const _Float16* __restrict__ Vt, int hRow, int ldbt,
                         _Float16* __restrict__ Cb, int cH, int ldc, int Nv) {
  int lane = threadIdx.x;
  int bh = blockIdx.x, b = bh / kH, h = bh % kH;
  int m0 = (blockIdx.y * 4 + threadIdx.y) * 16;
  int n0 = blockIdx.z * 16;
  int r = lane & 15, kb = (lane < 16) ? 0 : 8;
  const _Float16* Ar = attn + ((long)bh * kN + m0 + r) * kN + kb;
  int row = min(n0 + r, Nv - 1);
  const _Float16* Br = Vt + ((long)h * hRow + row) * ldbt + (long)b * kN + kb;
  v8f acc = {};
  for (int k = 0; k < kN; k += 32)
    acc = wmma32(fragH(Ar + k), fragH(Br + k), acc);
  int col = n0 + r, mb = (lane < 16) ? 0 : 8;
  if (col < Nv) {
    _Float16* C = Cb + ((long)b * kN) * ldc + cH * h;
#pragma unroll
    for (int q = 0; q < 8; ++q)
      C[(long)(m0 + mb + q) * ldc + col] = (_Float16)acc[q];
  }
}

// ---------------- res_pair: attn[b,:,i,:](12x512) @ pair[b,i](512x128) ------
// block (32,8) per (b,i). A slice staged in LDS once. Pairwise streamed as
// 32x128 f32 chunks: TDM double-buffered (wave0 issues chunk ci+1, waits
// TENSORcnt<=1 so chunk ci is complete, then all waves compute while the DMA
// streams the next buffer). Cooperative-copy fallback if TDM absent.
__global__ void k_res_pair(const _Float16* __restrict__ attn,
                           const float* __restrict__ pair,
                           _Float16* __restrict__ concatH) {
  __shared__ alignas(16) _Float16 As[16 * kN];          // 16 KB (rows 12..15 unused)
  __shared__ alignas(16) float Pf[2][32 * kPAIR];       // 2 x 16 KB ping-pong
  int lane = threadIdx.x, tid = threadIdx.y * 32 + lane;
  int bi = blockIdx.x, b = bi / kN, i = bi % kN;
  // stage attention slice: 12 rows x 512 halves = 768 int4 chunks
  for (int t = tid; t < kH * (kN / 8); t += 256) {
    int h = t >> 6, x = t & 63;
    ((int4*)As)[(long)h * 64 + x] =
        ((const int4*)(attn + (((long)b * kH + h) * kN + i) * kN))[x];
  }
  int n0 = threadIdx.y * 16;
  int r = lane & 15, kb = (lane < 16) ? 0 : 8;
  int c = n0 + r;
  const _Float16* Ar = As + (long)r * kN + kb;
  const float* P = pair + (long)bi * kN * kPAIR;
  constexpr int NC = kN / 32;                           // 16 chunks
  v8f acc = {};
#if HAVE_TDM
  unsigned lds0 = (unsigned)(unsigned long long)(const void*)&Pf[0][0];
  unsigned lds1 = (unsigned)(unsigned long long)(const void*)&Pf[1][0];
  if (threadIdx.y == 0)
    tdm_load_chunk(P, lds0);                            // prologue: chunk 0
#endif
  for (int ci = 0; ci < NC; ++ci) {
#if HAVE_TDM
    if (threadIdx.y == 0) {
      if (ci + 1 < NC) {
        tdm_load_chunk(P + (long)(ci + 1) * 32 * kPAIR, ((ci + 1) & 1) ? lds1 : lds0);
        __builtin_amdgcn_s_wait_tensorcnt(1);           // chunk ci done; ci+1 in flight
      } else {
        __builtin_amdgcn_s_wait_tensorcnt(0);
      }
    }
    const float* Pc = Pf[ci & 1];
#else
    const float* Pchunk = P + (long)ci * 32 * kPAIR;
    for (int t = tid; t < (32 * kPAIR) / 4; t += 256)
      ((int4*)Pf[0])[t] = ((const int4*)Pchunk)[t];
    const float* Pc = Pf[0];
#endif
    __syncthreads();
    v16h bf;
#pragma unroll
    for (int e = 0; e < 8; ++e) {
      bf[e]     = (_Float16)Pc[(kb + e) * kPAIR + c];
      bf[e + 8] = (_Float16)Pc[(kb + 16 + e) * kPAIR + c];
    }
    acc = wmma32(fragH(Ar + ci * 32), bf, acc);
    __syncthreads();                                    // done reading before overwrite
  }
  int mb = (lane < 16) ? 0 : 8;
#pragma unroll
  for (int q = 0; q < 8; ++q) {
    int h = mb + q;
    if (h < kH)
      concatH[(long)bi * kCOL + 576 + h * kPAIR + c] = (_Float16)acc[q];
  }
}

// ---------------- Rotate back res_p, norms -> concatH -----------------------
__global__ void k_rotate_back(const _Float16* __restrict__ rpH,
                              const float* __restrict__ rot,
                              const float* __restrict__ trans,
                              _Float16* __restrict__ concatH) {
  int idx = blockIdx.x * blockDim.x + threadIdx.x;
  if (idx >= kB * kN * kH * 8) return;
  int p = idx & 7, t1 = idx >> 3;
  int h = t1 % kH, bn = t1 / kH;
  const float* R = rot + (long)bn * 9;
  const float* t = trans + (long)bn * 3;
  const _Float16* v = rpH + (long)bn * 288 + h * 24 + p * 3;
  float x = (float)v[0] - t[0], y = (float)v[1] - t[1], z = (float)v[2] - t[2];
  float o0 = x*R[0] + y*R[1] + z*R[2];                 // out[r]=sum_c v[c]*R[r,c]
  float o1 = x*R[3] + y*R[4] + z*R[5];
  float o2 = x*R[6] + y*R[7] + z*R[8];
  _Float16* out = concatH + (long)bn * kCOL;
  out[192 + h*24 + p*3 + 0] = (_Float16)o0;
  out[192 + h*24 + p*3 + 1] = (_Float16)o1;
  out[192 + h*24 + p*3 + 2] = (_Float16)o2;
  out[480 + h*8 + p] = (_Float16)sqrtf(o0*o0 + o1*o1 + o2*o2 + kEps);
}

// ---------------------------------------------------------------------------
extern "C" void kernel_launch(void* const* d_in, const int* in_sizes, int n_in,
                              void* d_out, int out_size, void* d_ws, size_t ws_size,
                              hipStream_t stream) {
  const float* x      = (const float*)d_in[0];
  const float* pair   = (const float*)d_in[1];
  const float* rot    = (const float*)d_in[2];
  const float* trans  = (const float*)d_in[3];
  const float* W_sq   = (const float*)d_in[5];
  const float* W_sk   = (const float*)d_in[6];
  const float* W_sv   = (const float*)d_in[7];
  const float* W_pq   = (const float*)d_in[8];
  const float* W_pk   = (const float*)d_in[9];
  const float* W_pv   = (const float*)d_in[10];
  const float* pw_raw = (const float*)d_in[11];
  const float* W_pb   = (const float*)d_in[12];
  const float* b_pb   = (const float*)d_in[13];
  const float* W_out  = (const float*)d_in[14];
  const float* b_out  = (const float*)d_in[15];
  float* out = (float*)d_out;

  const long M = (long)kB * kN;                        // 1024
  char* ws = (char*)d_ws;
  size_t cur = 0;
  auto alloc = [&](size_t bytes) { void* p = ws + cur; cur += (bytes + 255) & ~(size_t)255; return p; };
  _Float16* xh    = (_Float16*)alloc(M * kDIM * 2);
  _Float16* wsqT  = (_Float16*)alloc(192 * kDIM * 2);
  _Float16* wskT  = (_Float16*)alloc(192 * kDIM * 2);
  _Float16* wsvT  = (_Float16*)alloc(192 * kDIM * 2);
  _Float16* wpqT  = (_Float16*)alloc(144 * kDIM * 2);
  _Float16* wpkT  = (_Float16*)alloc(144 * kDIM * 2);
  _Float16* wpvT  = (_Float16*)alloc(288 * kDIM * 2);
  _Float16* woutT = (_Float16*)alloc((long)kDIM * kCOL * 2);
  _Float16* wpbT  = (_Float16*)alloc(16 * kPAIR * 2);
  float*    sq    = (float*)alloc(M * 192 * 4);
  float*    sk    = (float*)alloc(M * 192 * 4);
  float*    pq    = (float*)alloc(M * 144 * 4);
  float*    pk    = (float*)alloc(M * 144 * 4);
  float*    pv    = (float*)alloc(M * 288 * 4);
  _Float16* svT   = (_Float16*)alloc(192 * M * 2);     // v_s transposed f16
  _Float16* vpT   = (_Float16*)alloc(288 * M * 2);     // v_p rotated, transposed
  _Float16* Apck  = (_Float16*)alloc((long)kB * kH * kN * 32 * 2);
  _Float16* Bpck  = (_Float16*)alloc((long)kB * kH * kN * 32 * 2);
  float*    rowA  = (float*)alloc((long)kB * kH * kN * 4);
  float*    colA  = (float*)alloc((long)kB * kH * kN * 4);
  float*    logits= (float*)alloc((long)kB * kH * kN * kN * 4);
  _Float16* attn  = (_Float16*)alloc((long)kB * kH * kN * kN * 2);
  _Float16* rpH   = (_Float16*)alloc(M * 288 * 2);
  _Float16* conc  = (_Float16*)alloc(M * kCOL * 2);

  dim3 wblk(32, 4);

  // 0) one-time f16 conversions / transposes
  hipLaunchKernelGGL(k_cvt_h16, dim3((M * kDIM + 255) / 256), dim3(256), 0, stream, x, xh, (int)(M * kDIM));
  hipLaunchKernelGGL(k_cvt_wt, dim3((192 * kDIM + 255) / 256), dim3(256), 0, stream, W_sq, kDIM, 192, wsqT, 192);
  hipLaunchKernelGGL(k_cvt_wt, dim3((192 * kDIM + 255) / 256), dim3(256), 0, stream, W_sk, kDIM, 192, wskT, 192);
  hipLaunchKernelGGL(k_cvt_wt, dim3((192 * kDIM + 255) / 256), dim3(256), 0, stream, W_sv, kDIM, 192, wsvT, 192);
  hipLaunchKernelGGL(k_cvt_wt, dim3((144 * kDIM + 255) / 256), dim3(256), 0, stream, W_pq, kDIM, 144, wpqT, 144);
  hipLaunchKernelGGL(k_cvt_wt, dim3((144 * kDIM + 255) / 256), dim3(256), 0, stream, W_pk, kDIM, 144, wpkT, 144);
  hipLaunchKernelGGL(k_cvt_wt, dim3((288 * kDIM + 255) / 256), dim3(256), 0, stream, W_pv, kDIM, 288, wpvT, 288);
  hipLaunchKernelGGL(k_cvt_wt, dim3(((long)kDIM * kCOL + 255) / 256), dim3(256), 0, stream, W_out, kCOL, kDIM, woutT, kDIM);
  hipLaunchKernelGGL(k_cvt_wt, dim3((16 * kPAIR + 255) / 256), dim3(256), 0, stream, W_pb, kPAIR, kH, wpbT, 16);

  // 1) projections (f16 WMMA, fp32 accum)
  hipLaunchKernelGGL(k_gemm_n, dim3(64, 3), wblk, 0, stream, xh, kDIM, wsqT, kDIM, sq, 192, kDIM, 192, (const float*)nullptr);
  hipLaunchKernelGGL(k_gemm_n, dim3(64, 3), wblk, 0, stream, xh, kDIM, wskT, kDIM, sk, 192, kDIM, 192, (const float*)nullptr);
  hipLaunchKernelGGL(k_gemm_n, dim3(64, 3), wblk, 0, stream, xh, kDIM, wpqT, kDIM, pq, 144, kDIM, 144, (const float*)nullptr);
  hipLaunchKernelGGL(k_gemm_n, dim3(64, 3), wblk, 0, stream, xh, kDIM, wpkT, kDIM, pk, 144, kDIM, 144, (const float*)nullptr);
  hipLaunchKernelGGL(k_gemm_n, dim3(64, 5), wblk, 0, stream, xh, kDIM, wpvT, kDIM, pv, 288, kDIM, 288, (const float*)nullptr);
  hipLaunchKernelGGL(k_gemm_t, dim3(64, 3), wblk, 0, stream, xh, kDIM, wsvT, kDIM, svT, (int)M, kDIM, 192);

  // 2) rotate + pack
  hipLaunchKernelGGL(k_rotate_pack, dim3((kB * kN * kH + 127) / 128), dim3(128), 0, stream,
                     sq, sk, pq, pk, pv, rot, trans, pw_raw, Apck, Bpck, rowA, colA, vpT);

  // 3) pair bias -> logits ; 4) QK+point WMMA accumulates on top
  hipLaunchKernelGGL(k_pair_bias, dim3(kB * kN, 8), wblk, 0, stream, pair, wpbT, b_pb, logits);
  hipLaunchKernelGGL(k_qk_logits, dim3(kB * kH, 32, 8), wblk, 0, stream, Apck, Bpck, rowA, colA, logits);

  // 5) softmax
  hipLaunchKernelGGL(k_softmax, dim3(kB * kH * kN / 4), wblk, 0, stream, logits, attn);

  // 6) attn @ v_s -> concat[:,0:192] ; attn @ v_p -> rpH
  hipLaunchKernelGGL(k_attn_v, dim3(kB * kH, 8, 1), wblk, 0, stream, attn, svT, 16, (int)M, conc, 16, kCOL, 16);
  hipLaunchKernelGGL(k_attn_v, dim3(kB * kH, 8, 2), wblk, 0, stream, attn, vpT, 24, (int)M, rpH, 24, 288, 24);

  // 7) attn @ pairwise -> concat[:,576:2112]   (TDM double-buffered)
  hipLaunchKernelGGL(k_res_pair, dim3(kB * kN), dim3(32, 8), 0, stream, attn, pair, conc);

  // 8) rotate back + norms -> concat[:,192:576]
  hipLaunchKernelGGL(k_rotate_back, dim3((kB * kN * kH * 8 + 255) / 256), dim3(256), 0, stream,
                     rpH, rot, trans, conc);

  // 9) output projection (+bias) -> d_out
  hipLaunchKernelGGL(k_gemm_n, dim3(64, 6), wblk, 0, stream, conc, kCOL, woutT, kCOL, out, kDIM, kCOL, kDIM, b_out);
}
